// HybridQwen25ForCausalLM_7009386627436
// MI455X (gfx1250) — matrix operand order repaired and checked by reading, not compiled, not run on hardware
//
#include <hip/hip_runtime.h>
#include <hip/hip_bf16.h>

// ---------------------------------------------------------------------------
// Qwen2.5-style 4-layer decoder forward for MI455X (gfx1250, wave32, WMMA).
// f32 global -> bf16 LDS tiles (fragment-friendly layouts, b128 DS loads)
// -> v_wmma_f32_16x16x32_bf16 -> f32. Double-buffered LDS pipelines in the
// GEMM and attention hot loops; weight streams prefetched two tiles ahead.
// ---------------------------------------------------------------------------

#define L_   4
#define S_   2048
#define V_   32000
#define H_   2048
#define NH_  16
#define NKV_ 4
#define HD_  128
#define FF_  8192
#define KVW_ (NKV_ * HD_)   // 512
#define EPS_ 1e-6f

typedef __attribute__((ext_vector_type(16))) __bf16 v16bf;
typedef __attribute__((ext_vector_type(4)))  __bf16 v4bf;
typedef __attribute__((ext_vector_type(8)))  float  v8f;

static __device__ inline v8f v8f_zero() {
    v8f z;
#pragma unroll
    for (int i = 0; i < 8; ++i) z[i] = 0.0f;
    return z;
}

// A-matrix (16x32 bf16) element j -> K index for this lane (ISA 7.12.2).
// Contiguous for j=0..7 and j=8..15 -> two 16B runs -> ds_load_b128 x2.
static __device__ inline int kA(int lane, int j) {
    return (j & 7) + ((j >> 3) << 4) + ((lane >> 4) << 3);
}
// B-matrix (32x16 bf16) element j -> contraction index. Same order as kA.
static __device__ inline int kB(int lane, int j) {
    return (j & 7) + ((j >> 3) << 4) + ((lane >> 4) << 3);
}

static __device__ inline v4bf cvt4(const float* __restrict__ p) {
    float4 f = *(const float4*)p;
    v4bf t;
    t[0] = (__bf16)f.x; t[1] = (__bf16)f.y; t[2] = (__bf16)f.z; t[3] = (__bf16)f.w;
    return t;
}

// ---------------------------------------------------------------------------
// Tiled GEMM: C[M,N] = A[M,K] @ W[K,N] (+ bias[N]) (+ resid[M,N])
// 256 threads (8 waves). Block tile 128x128, wave tile 16x128, k-step 32.
// Double-buffered LDS: tile i+1 streams in while tile i feeds the WMMAs;
// one barrier per k-tile.
// ---------------------------------------------------------------------------
static __device__ inline void gemm_load_tiles(
    const float* __restrict__ A, const float* __restrict__ W,
    __bf16* __restrict__ la, __bf16* __restrict__ lb,
    int bm, int bn, int k0, int N, int K, int tid, int nB, int kg)
{
    // A tile 128x32: float4 global loads, packed b64 LDS stores.
#pragma unroll
    for (int i = 0; i < 4; ++i) {
        int idx = tid * 4 + i * 1024;            // 4096 elems
        int r = idx >> 5, c = idx & 31;
        *(v4bf*)&la[idx] = cvt4(&A[(size_t)(bm + r) * K + (k0 + c)]);
    }
    // B tile 32x128 -> transposed LDS [n][k]; coalesced b32 global loads.
#pragma unroll
    for (int i = 0; i < 4; ++i) {
        int kb4 = (kg + 2 * i) * 4;              // 0,8,16,24 / 4,12,20,28
        v4bf t;
#pragma unroll
        for (int ii = 0; ii < 4; ++ii)
            t[ii] = (__bf16)W[(size_t)(k0 + kb4 + ii) * N + (bn + nB)];
        *(v4bf*)&lb[nB * 32 + kb4] = t;
    }
}

template <bool HAS_BIAS, bool HAS_RESID>
__global__ __launch_bounds__(256) void gemm_wmma_kernel(
    const float* __restrict__ A, const float* __restrict__ W,
    const float* __restrict__ bias, const float* __restrict__ resid,
    float* __restrict__ C, int M, int N, int K)
{
    __shared__ __bf16 lds_a[2][128 * 32];   // 2 x 8 KB, [m][k]
    __shared__ __bf16 lds_b[2][128 * 32];   // 2 x 8 KB, [n][k] (transposed)

    const int tid  = threadIdx.x;
    const int lane = tid & 31;
    const int wid  = tid >> 5;           // 0..7
    const int bm   = blockIdx.y * 128;
    const int bn   = blockIdx.x * 128;
    const int nB   = tid & 127;          // B-tile column this thread fills
    const int kg   = tid >> 7;           // 0/1: which k sub-group

    v8f acc[8];
#pragma unroll
    for (int nt = 0; nt < 8; ++nt) acc[nt] = v8f_zero();

    // Prologue: fill buffer 0.
    gemm_load_tiles(A, W, lds_a[0], lds_b[0], bm, bn, 0, N, K, tid, nB, kg);
    __syncthreads();

    int buf = 0;
    for (int k0 = 0; k0 < K; k0 += 32, buf ^= 1) {
        // Stream next tile into the other buffer (overlaps with WMMAs below).
        if (k0 + 32 < K)
            gemm_load_tiles(A, W, lds_a[buf ^ 1], lds_b[buf ^ 1],
                            bm, bn, k0 + 32, N, K, tid, nB, kg);
        // Prefetch two tiles ahead (global_prefetch_b8).
        if (k0 + 64 < K) {
            int r = (tid * 4) >> 5, c = (tid * 4) & 31;
            __builtin_prefetch(&A[(size_t)(bm + r) * K + (k0 + 64 + c)], 0, 3);
            __builtin_prefetch(&W[(size_t)(k0 + 64 + kg * 4) * N + (bn + nB)], 0, 3);
        }

        const __bf16* la = lds_a[buf];
        const __bf16* lb = lds_b[buf];
        v16bf afrag;
        const int m = 16 * wid + (lane & 15);
#pragma unroll
        for (int j = 0; j < 16; ++j) afrag[j] = la[m * 32 + kA(lane, j)];

#pragma unroll
        for (int nt = 0; nt < 8; ++nt) {
            v16bf bfrag;
            const int n = 16 * nt + (lane & 15);
#pragma unroll
            for (int j = 0; j < 16; ++j) bfrag[j] = lb[n * 32 + kB(lane, j)];
            acc[nt] = __builtin_amdgcn_wmma_f32_16x16x32_bf16(
                false, afrag, false, bfrag, (short)0, acc[nt], false, false);
        }
        __syncthreads();   // next iter may overwrite buf; all reads done
    }

    // Epilogue. Row base once per j; 8 column stores differ by constant 64B.
    const int half = lane >> 4, nn = lane & 15;
    float bvals[8];
    if (HAS_BIAS) {
#pragma unroll
        for (int nt = 0; nt < 8; ++nt) bvals[nt] = bias[bn + 16 * nt + nn];
    }
#pragma unroll
    for (int j = 0; j < 8; ++j) {
        const int row = bm + 16 * wid + j + 8 * half;
        const size_t rbase = (size_t)row * N + (bn + nn);
        float* __restrict__ cp = C + rbase;
        const float* __restrict__ rp = HAS_RESID ? (resid + rbase) : nullptr;
#pragma unroll
        for (int nt = 0; nt < 8; ++nt) {
            float val = acc[nt][j];
            if (HAS_BIAS)  val += bvals[nt];
            if (HAS_RESID) val += rp[16 * nt];
            cp[16 * nt] = val;
        }
    }
}

// ---------------------------------------------------------------------------
// Embedding gather: h[s,:] = embed[ids[s],:]
// ---------------------------------------------------------------------------
__global__ void embed_kernel(const int* __restrict__ ids,
                             const float* __restrict__ embed,
                             float* __restrict__ h)
{
    int s = blockIdx.x;
    int t = ids[s];
    for (int i = threadIdx.x; i < H_; i += 256)
        h[(size_t)s * H_ + i] = embed[(size_t)t * H_ + i];
}

// ---------------------------------------------------------------------------
// RMSNorm: y = x * rsqrt(mean(x^2)+eps) * w   (one block per row)
// ---------------------------------------------------------------------------
__global__ void rmsnorm_kernel(const float* __restrict__ x,
                               const float* __restrict__ w,
                               float* __restrict__ y, int width)
{
    __shared__ float red[256];
    int row = blockIdx.x;
    const float* xr = x + (size_t)row * width;
    float s = 0.0f;
    for (int i = threadIdx.x; i < width; i += 256) { float t = xr[i]; s += t * t; }
    red[threadIdx.x] = s;
    __syncthreads();
    for (int off = 128; off > 0; off >>= 1) {
        if (threadIdx.x < off) red[threadIdx.x] += red[threadIdx.x + off];
        __syncthreads();
    }
    float inv = rsqrtf(red[0] / (float)width + EPS_);
    for (int i = threadIdx.x; i < width; i += 256)
        y[(size_t)row * width + i] = xr[i] * inv * w[i];
}

// ---------------------------------------------------------------------------
// RoPE (in place) on [S, nheads, 128]; grid (S, nheads), 64 threads (d=0..63)
// ---------------------------------------------------------------------------
__global__ void rope_kernel(float* __restrict__ x, int nheads)
{
    int s = blockIdx.x, hidx = blockIdx.y, d = threadIdx.x;
    float inv = __powf(1000000.0f, -(float)d / 64.0f);
    float f = (float)s * inv;
    float c = __cosf(f), sn = __sinf(f);
    float* p = x + ((size_t)s * nheads + hidx) * HD_;
    float x0 = p[d], x1 = p[d + 64];
    p[d]      = x0 * c - x1 * sn;
    p[d + 64] = x1 * c + x0 * sn;
}

// ---------------------------------------------------------------------------
// SiLU-gate: g = silu(g) * u
// ---------------------------------------------------------------------------
__global__ void silu_mul_kernel(float* __restrict__ g,
                                const float* __restrict__ u, size_t n)
{
    size_t i = (size_t)blockIdx.x * blockDim.x + threadIdx.x;
    if (i < n) {
        float x = g[i];
        g[i] = (x / (1.0f + __expf(-x))) * u[i];
    }
}

// ---------------------------------------------------------------------------
// Flash attention, causal, GQA (16 q-heads, 4 kv-heads).
// 128 threads (4 waves): one head x 64 query rows per block, 16 q per wave.
// Key blocks of 32, double-buffered K|Vt LDS. K row-major (K-frag
// contiguous); V transposed [d][key] (V-frag contiguous); P round-trips
// through LDS to convert C-layout -> A-fragment layout.
// ---------------------------------------------------------------------------
static __device__ inline void attn_load_kv(
    const float* __restrict__ k, const float* __restrict__ v,
    __bf16* __restrict__ buf, int kbase, int kvh, int tid)
{
    // K 32x128 row-major into buf[0..4095].
#pragma unroll
    for (int i = 0; i < 8; ++i) {
        int idx = tid * 4 + i * 512;
        int r = idx >> 7, c = idx & 127;
        *(v4bf*)&buf[idx] = cvt4(&k[((size_t)(kbase + r) * NKV_ + kvh) * HD_ + c]);
    }
    // V transposed [d][key] into buf[4096..8191].
    const int d = tid;                    // 0..127
#pragma unroll
    for (int i = 0; i < 8; ++i) {
        int r4 = i * 4;
        v4bf t;
#pragma unroll
        for (int ii = 0; ii < 4; ++ii)
            t[ii] = (__bf16)v[((size_t)(kbase + r4 + ii) * NKV_ + kvh) * HD_ + d];
        *(v4bf*)&buf[4096 + d * 32 + r4] = t;
    }
}

__global__ __launch_bounds__(128) void attn_kernel(
    const float* __restrict__ q, const float* __restrict__ k,
    const float* __restrict__ v, float* __restrict__ out)
{
    __shared__ __bf16 ldsKV[2][8192];  // per buffer: K(32x128) | Vt(128x32)
    __shared__ __bf16 ldsP[2048];      // per-wave 16x32 probability tiles

    const int   hh   = blockIdx.x;
    const int   q0   = blockIdx.y * 64;
    const int   kvh  = hh >> 2;               // NH/NKV = 4
    const int   tid  = threadIdx.x;
    const int   lane = tid & 31;
    const int   wid  = tid >> 5;
    const int   half = lane >> 4, nn = lane & 15;
    const float scale = 0.08838834764831845f; // 1/sqrt(128)

    // Stage Q tile (64x128) through buffer 0; build per-wave A-fragments.
#pragma unroll
    for (int i = 0; i < 16; ++i) {
        int idx = tid * 4 + i * 512;
        int r = idx >> 7, c = idx & 127;
        *(v4bf*)&ldsKV[0][idx] = cvt4(&q[((size_t)(q0 + r) * NH_ + hh) * HD_ + c]);
    }
    __syncthreads();

    v16bf qfrag[4];
    {
        const int m = lane & 15;
#pragma unroll
        for (int kc = 0; kc < 4; ++kc)
#pragma unroll
            for (int j = 0; j < 16; ++j)
                qfrag[kc][j] = ldsKV[0][(16 * wid + m) * 128 + 32 * kc + kA(lane, j)];
    }
    __syncthreads();

    v8f oacc[8];
#pragma unroll
    for (int c = 0; c < 8; ++c) oacc[c] = v8f_zero();
    float mrow[8], lrow[8];
#pragma unroll
    for (int j = 0; j < 8; ++j) { mrow[j] = -1e30f; lrow[j] = 0.0f; }

    const int nkb = (q0 >> 5) + 2; // key blocks covering keys <= q0+63

    // Prologue: fill buffer 0 with key block 0.
    attn_load_kv(k, v, ldsKV[0], 0, kvh, tid);
    __syncthreads();

    int buf = 0;
    for (int kb = 0; kb < nkb; ++kb, buf ^= 1) {
        const int kbase = kb * 32;
        // Stream next key block into the other buffer.
        if (kb + 1 < nkb)
            attn_load_kv(k, v, ldsKV[buf ^ 1], kbase + 32, kvh, tid);

        const __bf16* ldsK  = ldsKV[buf];          // [key][d]  32x128
        const __bf16* ldsVt = ldsKV[buf] + 4096;   // [d][key]  128x32

#pragma unroll
        for (int t = 0; t < 2; ++t) {
            // scores: S(16q x 16k) = Q(16x128) * K^T, chunked over head dim.
            v8f sacc = v8f_zero();
#pragma unroll
            for (int kc = 0; kc < 4; ++kc) {
                v16bf bfrag;
#pragma unroll
                for (int j = 0; j < 16; ++j)
                    bfrag[j] = ldsK[(16 * t + nn) * 128 + 32 * kc + kB(lane, j)];
                sacc = __builtin_amdgcn_wmma_f32_16x16x32_bf16(
                    false, qfrag[kc], false, bfrag, (short)0, sacc, false, false);
            }
            // scale + causal mask; online softmax (row = j + 8*half).
            const int kg = kbase + 16 * t + nn;
#pragma unroll
            for (int j = 0; j < 8; ++j) {
                int qg = q0 + 16 * wid + j + 8 * half;
                sacc[j] = sacc[j] * scale + (kg <= qg ? 0.0f : -1e9f);
            }
#pragma unroll
            for (int j = 0; j < 8; ++j) {
                float m2 = sacc[j];
#pragma unroll
                for (int d = 1; d < 16; d <<= 1)
                    m2 = fmaxf(m2, __shfl_xor(m2, d, 32));
                float newm = fmaxf(mrow[j], m2);
                float fac  = __expf(mrow[j] - newm);
                float p    = __expf(sacc[j] - newm);
                float ps   = p;
#pragma unroll
                for (int d = 1; d < 16; d <<= 1)
                    ps += __shfl_xor(ps, d, 32);
                lrow[j] = lrow[j] * fac + ps;
                mrow[j] = newm;
                sacc[j] = p;
#pragma unroll
                for (int c = 0; c < 8; ++c) oacc[c][j] *= fac;
            }
            // Stash P tile (C-layout) into this wave's 16x32 LDS region.
#pragma unroll
            for (int j = 0; j < 8; ++j)
                ldsP[wid * 512 + (j + 8 * half) * 32 + 16 * t + nn] = (__bf16)sacc[j];
        }

        // Re-read P as 16x32 A-fragment (contiguous b128 runs); O += P*V.
        v16bf pfrag;
        {
            const int m = lane & 15;
#pragma unroll
            for (int j = 0; j < 16; ++j)
                pfrag[j] = ldsP[wid * 512 + m * 32 + kA(lane, j)];
        }
#pragma unroll
        for (int c = 0; c < 8; ++c) {
            v16bf vfrag;
#pragma unroll
            for (int j = 0; j < 16; ++j)
                vfrag[j] = ldsVt[(16 * c + nn) * 32 + kB(lane, j)];
            oacc[c] = __builtin_amdgcn_wmma_f32_16x16x32_bf16(
                false, pfrag, false, vfrag, (short)0, oacc[c], false, false);
        }
        __syncthreads();   // next iter may overwrite this buffer
    }

    // Normalize and write. Per-row base; column stores differ by 64B.
#pragma unroll
    for (int j = 0; j < 8; ++j) {
        const int row = q0 + 16 * wid + j + 8 * half;
        float* __restrict__ op = out + ((size_t)row * NH_ + hh) * HD_ + nn;
        const float rinv = 1.0f / lrow[j];
#pragma unroll
        for (int c = 0; c < 8; ++c)
            op[16 * c] = oacc[c][j] * rinv;
    }
}

// ---------------------------------------------------------------------------
// Host-side orchestration
// ---------------------------------------------------------------------------
extern "C" void kernel_launch(void* const* d_in, const int* in_sizes, int n_in,
                              void* d_out, int out_size, void* d_ws, size_t ws_size,
                              hipStream_t stream)
{
    const int*   ids   = (const int*)  d_in[0];
    const float* embed = (const float*)d_in[1];
    const float* Wq    = (const float*)d_in[2];
    const float* bq    = (const float*)d_in[3];
    const float* Wk    = (const float*)d_in[4];
    const float* bk    = (const float*)d_in[5];
    const float* Wv    = (const float*)d_in[6];
    const float* bv    = (const float*)d_in[7];
    const float* Wo    = (const float*)d_in[8];
    const float* ln1   = (const float*)d_in[9];
    const float* ln2   = (const float*)d_in[10];
    const float* Wg    = (const float*)d_in[11];
    const float* Wu    = (const float*)d_in[12];
    const float* Wd    = (const float*)d_in[13];
    const float* normw = (const float*)d_in[14];
    const float* Wlm   = (const float*)d_in[15];
    const float* blm   = (const float*)d_in[16];
    float* outp = (float*)d_out;

    const size_t SH = (size_t)S_ * H_;
    float* ws   = (float*)d_ws;
    float* h    = ws;
    float* xn   = ws + SH;
    float* qb   = ws + 2 * SH;
    float* attn = ws + 3 * SH;
    float* kb_  = ws + 4 * SH;
    float* vb   = ws + 4 * SH + (size_t)S_ * KVW_;
    float* gb   = ws + 4 * SH + 2 * (size_t)S_ * KVW_;
    float* ub   = gb + (size_t)S_ * FF_;

    embed_kernel<<<S_, 256, 0, stream>>>(ids, embed, h);

    for (int l = 0; l < L_; ++l) {
        const float* Wql = Wq + (size_t)l * H_ * H_;
        const float* Wkl = Wk + (size_t)l * H_ * KVW_;
        const float* Wvl = Wv + (size_t)l * H_ * KVW_;
        const float* Wol = Wo + (size_t)l * H_ * H_;
        const float* Wgl = Wg + (size_t)l * H_ * FF_;
        const float* Wul = Wu + (size_t)l * H_ * FF_;
        const float* Wdl = Wd + (size_t)l * FF_ * H_;

        rmsnorm_kernel<<<S_, 256, 0, stream>>>(h, ln1 + (size_t)l * H_, xn, H_);

        gemm_wmma_kernel<true, false><<<dim3(H_ / 128, S_ / 128), 256, 0, stream>>>(
            xn, Wql, bq + (size_t)l * H_, nullptr, qb, S_, H_, H_);
        gemm_wmma_kernel<true, false><<<dim3(KVW_ / 128, S_ / 128), 256, 0, stream>>>(
            xn, Wkl, bk + (size_t)l * KVW_, nullptr, kb_, S_, KVW_, H_);
        gemm_wmma_kernel<true, false><<<dim3(KVW_ / 128, S_ / 128), 256, 0, stream>>>(
            xn, Wvl, bv + (size_t)l * KVW_, nullptr, vb, S_, KVW_, H_);

        rope_kernel<<<dim3(S_, NH_),  64, 0, stream>>>(qb, NH_);
        rope_kernel<<<dim3(S_, NKV_), 64, 0, stream>>>(kb_, NKV_);

        attn_kernel<<<dim3(NH_, S_ / 64), 128, 0, stream>>>(qb, kb_, vb, attn);

        gemm_wmma_kernel<false, true><<<dim3(H_ / 128, S_ / 128), 256, 0, stream>>>(
            attn, Wol, nullptr, h, h, S_, H_, H_);

        rmsnorm_kernel<<<S_, 256, 0, stream>>>(h, ln2 + (size_t)l * H_, xn, H_);

        gemm_wmma_kernel<false, false><<<dim3(FF_ / 128, S_ / 128), 256, 0, stream>>>(
            xn, Wgl, nullptr, nullptr, gb, S_, FF_, H_);
        gemm_wmma_kernel<false, false><<<dim3(FF_ / 128, S_ / 128), 256, 0, stream>>>(
            xn, Wul, nullptr, nullptr, ub, S_, FF_, H_);

        size_t nff = (size_t)S_ * FF_;
        silu_mul_kernel<<<(unsigned)((nff + 255) / 256), 256, 0, stream>>>(gb, ub, nff);

        gemm_wmma_kernel<false, true><<<dim3(H_ / 128, S_ / 128), 256, 0, stream>>>(
            gb, Wdl, nullptr, h, h, S_, H_, FF_);
    }

    rmsnorm_kernel<<<S_, 256, 0, stream>>>(h, normw, xn, H_);
    gemm_wmma_kernel<true, false><<<dim3(V_ / 128, S_ / 128), 256, 0, stream>>>(
        xn, Wlm, blm, nullptr, outp, S_, V_, H_);
}
